// LayerWiseSelfAttentionPooler_63668595195997
// MI455X (gfx1250) — compile-verified, matrix operand order checked by reading, not run on hardware
//
#include <hip/hip_runtime.h>

// ---------------------------------------------------------------------------
// LayerWiseSelfAttentionPooler for MI455X (gfx1250, wave32, WMMA bf16)
// All heavy GEMMs + attention use v_wmma_f32_16x16x32_bf16 with fp32 accum.
// ---------------------------------------------------------------------------

typedef __attribute__((ext_vector_type(16))) __bf16 v16bf;
typedef __attribute__((ext_vector_type(8)))  __bf16 bf16x8;
typedef __attribute__((ext_vector_type(8)))  float  v8f;

#define LAYERS 12
#define Bsz    8
#define Ssz    512
#define Esz    768
#define Hsz    12
#define DHsz   64
#define QKVN   2304   // 3*E

__device__ __forceinline__ v16bf combine8(bf16x8 lo, bf16x8 hi) {
  v16bf r;
#pragma unroll
  for (int i = 0; i < 8; i++) { r[i] = lo[i]; r[i + 8] = hi[i]; }
  return r;
}

__device__ __forceinline__ v8f wmma_bf16(v16bf a, v16bf b, v8f c) {
  return __builtin_amdgcn_wmma_f32_16x16x32_bf16(false, a, false, b, (short)0, c,
                                                 false, false);
}

// ---------------------------------------------------------------------------
// fp32 -> bf16 cast
// ---------------------------------------------------------------------------
__global__ __launch_bounds__(256) void cast_bf16(const float* __restrict__ src,
                                                 __bf16* __restrict__ dst, int n) {
  int i = blockIdx.x * 256 + threadIdx.x;
  if (i < n) dst[i] = (__bf16)src[i];
}

// ---------------------------------------------------------------------------
// GEMM: out[M,N] = act(A[M,K] * W[N,K]^T + bias[N])
// FLAGS (compile time): 1 = ReLU, 2 = accumulate into fp32 out, 4 = bf16 out
// block = 128 threads (4 waves), tile 64x64, K-step 32 staged in LDS,
// software-pipelined global loads overlapping the WMMA compute.
// ---------------------------------------------------------------------------
template <int FLAGS>
__global__ __launch_bounds__(128) void gemm_wmma(
    const __bf16* __restrict__ A, const __bf16* __restrict__ W,
    const float* __restrict__ bias, float* __restrict__ outF,
    __bf16* __restrict__ outB, int M, int N, int K) {
  __shared__ __bf16 As[64][40];   // pad to 40 -> 80B row stride, conflict-free b128
  __shared__ __bf16 Bs[64][40];

  const int tid  = threadIdx.x;
  const int lane = tid & 31;
  const int wave = tid >> 5;
  const int m0 = blockIdx.x * 64;
  const int n0 = blockIdx.y * 64;
  const int wr = (wave >> 1) * 32;       // wave sub-tile row
  const int wc = (wave & 1) * 32;        // wave sub-tile col
  const int lrow = tid >> 1;             // LDS-fill row 0..63
  const int lcol = (tid & 1) * 16;       // LDS-fill col half (16 bf16 = 32B)
  const int fm  = lane & 15;
  const int kh8 = (lane >> 4) * 8;       // K sub-pattern per half-wave

  const __bf16* gA = A + (size_t)(m0 + lrow) * K + lcol;
  const __bf16* gB = W + (size_t)(n0 + lrow) * K + lcol;

  v8f c[2][2] = {};

  // prologue: fetch first K-tile
  uint4 a0 = ((const uint4*)gA)[0], a1 = ((const uint4*)gA)[1];
  uint4 b0 = ((const uint4*)gB)[0], b1 = ((const uint4*)gB)[1];

  for (int k0 = 0; k0 < K; k0 += 32) {
    __syncthreads();   // previous iteration's fragment reads complete
    *(uint4*)&As[lrow][lcol]     = a0;
    *(uint4*)&As[lrow][lcol + 8] = a1;
    *(uint4*)&Bs[lrow][lcol]     = b0;
    *(uint4*)&Bs[lrow][lcol + 8] = b1;
    __syncthreads();

    // issue next tile's global loads now; latency hides under the WMMAs below
    if (k0 + 32 < K) {
      const uint4* ga = (const uint4*)(gA + k0 + 32);
      const uint4* gb = (const uint4*)(gB + k0 + 32);
      a0 = ga[0]; a1 = ga[1];
      b0 = gb[0]; b1 = gb[1];
    }

    v16bf bfrag[2];
#pragma unroll
    for (int nt = 0; nt < 2; nt++)
      bfrag[nt] = combine8(*(const bf16x8*)&Bs[wc + nt * 16 + fm][kh8],
                           *(const bf16x8*)&Bs[wc + nt * 16 + fm][16 + kh8]);
#pragma unroll
    for (int mt = 0; mt < 2; mt++) {
      v16bf afrag = combine8(*(const bf16x8*)&As[wr + mt * 16 + fm][kh8],
                             *(const bf16x8*)&As[wr + mt * 16 + fm][16 + kh8]);
#pragma unroll
      for (int nt = 0; nt < 2; nt++)
        c[mt][nt] = wmma_bf16(afrag, bfrag[nt], c[mt][nt]);
    }
  }

  const int rowoff = (lane >> 4) * 8;  // C layout: lanes 16-31 hold rows M+8
#pragma unroll
  for (int mt = 0; mt < 2; mt++) {
#pragma unroll
    for (int nt = 0; nt < 2; nt++) {
      const int col = n0 + wc + nt * 16 + fm;
      const float bcol = bias[col];
      size_t idx = (size_t)(m0 + wr + mt * 16 + rowoff) * N + col;
#pragma unroll
      for (int r = 0; r < 8; r++) {
        float v = c[mt][nt][r] + bcol;
        if (FLAGS & 1) v = fmaxf(v, 0.0f);
        if (FLAGS & 4)      outB[idx] = (__bf16)v;
        else if (FLAGS & 2) outF[idx] += v;
        else                outF[idx] = v;
        idx += (size_t)N;
      }
    }
  }
}

// ---------------------------------------------------------------------------
// Fused self-attention: one wave per (b, h, 32-query tile).
// QK^T (WMMA) -> scores in 64KB LDS (column-major [key][q]) -> softmax
// (per-wave, in-order LDS) -> P.V (WMMA) -> bf16 context [b*S+q][h*64+d].
// scores get "+ mask_q*mask_k" additive term exactly as reference.
// ---------------------------------------------------------------------------
__global__ __launch_bounds__(32) void attn_wmma(
    const __bf16* __restrict__ qkv, const int* __restrict__ mask,
    __bf16* __restrict__ ctx) {
  __shared__ float sc[Ssz][32];   // 64 KB

  const int lane = threadIdx.x;
  const int blk  = blockIdx.x;
  const int qt = blk & 15;                 // S/32 = 16 query tiles
  const int h  = (blk >> 4) % Hsz;
  const int b  = blk / (16 * Hsz);
  const int q0 = qt * 32;
  const int fm  = lane & 15;
  const int kh8 = (lane >> 4) * 8;
  const int rowoff = (lane >> 4) * 8;

  // Q fragments resident in registers: 2 m-subtiles x 2 k-steps (d 0..63)
  v16bf aq[2][2];
#pragma unroll
  for (int mt = 0; mt < 2; mt++)
#pragma unroll
    for (int ks = 0; ks < 2; ks++) {
      const __bf16* qp =
          qkv + ((size_t)(b * Ssz + q0 + mt * 16 + fm)) * QKVN + h * DHsz + ks * 32;
      aq[mt][ks] = combine8(*(const bf16x8*)(qp + kh8),
                            *(const bf16x8*)(qp + 16 + kh8));
    }

  // hoist query-side mask values (invariant over key chunks)
  float mqv[2][8];
#pragma unroll
  for (int mt = 0; mt < 2; mt++)
#pragma unroll
    for (int r = 0; r < 8; r++)
      mqv[mt][r] = (float)mask[b * Ssz + q0 + mt * 16 + r + rowoff];

  // ---- scores = Q K^T / 8 + mq*mk ----
  for (int kc = 0; kc < 16; kc++) {
    v8f s[2][2] = {};
#pragma unroll
    for (int ks = 0; ks < 2; ks++) {
      v16bf bk[2];
#pragma unroll
      for (int nt = 0; nt < 2; nt++) {
        const __bf16* kp = qkv +
            ((size_t)(b * Ssz + kc * 32 + nt * 16 + fm)) * QKVN + Esz + h * DHsz +
            ks * 32;
        bk[nt] = combine8(*(const bf16x8*)(kp + kh8),
                          *(const bf16x8*)(kp + 16 + kh8));
      }
#pragma unroll
      for (int mt = 0; mt < 2; mt++)
#pragma unroll
        for (int nt = 0; nt < 2; nt++)
          s[mt][nt] = wmma_bf16(aq[mt][ks], bk[nt], s[mt][nt]);
    }
#pragma unroll
    for (int nt = 0; nt < 2; nt++) {
      const int k = kc * 32 + nt * 16 + fm;
      const float mk = (float)mask[b * Ssz + k];
#pragma unroll
      for (int mt = 0; mt < 2; mt++)
#pragma unroll
        for (int r = 0; r < 8; r++)
          sc[k][mt * 16 + r + rowoff] = s[mt][nt][r] * 0.125f + mqv[mt][r] * mk;
    }
  }
  __syncthreads();

  // ---- row softmax: lane r owns query row r (column scan is conflict-free) ----
  {
    int r = lane;
    float mx = -1e30f;
    for (int k = 0; k < Ssz; k++) mx = fmaxf(mx, sc[k][r]);
    float sum = 0.f;
    for (int k = 0; k < Ssz; k++) {
      float e = __expf(sc[k][r] - mx);
      sc[k][r] = e;
      sum += e;
    }
    float inv = 1.0f / sum;
    for (int k = 0; k < Ssz; k++) sc[k][r] *= inv;
  }
  __syncthreads();

  // ---- O = P V ----
  v8f o[2][4] = {};
  for (int kc = 0; kc < 16; kc++) {
    v16bf pa[2];
#pragma unroll
    for (int mt = 0; mt < 2; mt++) {
      v16bf p;
#pragma unroll
      for (int i = 0; i < 16; i++) {
        int kk = ((i >> 3) << 4) + kh8 + (i & 7);
        p[i] = (__bf16)sc[kc * 32 + kk][mt * 16 + fm];
      }
      pa[mt] = p;
    }
#pragma unroll
    for (int nt = 0; nt < 4; nt++) {
      v16bf vb;
#pragma unroll
      for (int i = 0; i < 16; i++) {
        int kk = ((i >> 3) << 4) + kh8 + (i & 7);
        vb[i] = qkv[((size_t)(b * Ssz + kc * 32 + kk)) * QKVN + 2 * Esz + h * DHsz +
                    nt * 16 + fm];
      }
#pragma unroll
      for (int mt = 0; mt < 2; mt++)
        o[mt][nt] = wmma_bf16(pa[mt], vb, o[mt][nt]);
    }
  }
#pragma unroll
  for (int mt = 0; mt < 2; mt++)
#pragma unroll
    for (int nt = 0; nt < 4; nt++)
#pragma unroll
      for (int r = 0; r < 8; r++) {
        int q = q0 + mt * 16 + r + rowoff;
        int d = nt * 16 + fm;
        ctx[((size_t)(b * Ssz + q)) * Esz + h * DHsz + d] = (__bf16)o[mt][nt][r];
      }
}

// ---------------------------------------------------------------------------
// seq = acc/12 ; emit bf16 seq and bf16 relu(seq)
// ---------------------------------------------------------------------------
__global__ __launch_bounds__(256) void seq_kernel(const float* __restrict__ acc,
                                                  __bf16* __restrict__ seqb,
                                                  __bf16* __restrict__ rseqb,
                                                  int n) {
  int i = blockIdx.x * 256 + threadIdx.x;
  if (i < n) {
    float s = acc[i] * (1.0f / 12.0f);
    seqb[i]  = (__bf16)s;
    rseqb[i] = (__bf16)fmaxf(s, 0.0f);
  }
}

// q = ones @ wq^T + bq  ->  q[e] = rowsum(fin_in_proj_w[e]) + b[e], e < E
__global__ __launch_bounds__(256) void qvec_kernel(const float* __restrict__ w,
                                                   const float* __restrict__ bias,
                                                   float* __restrict__ qv) {
  int e = blockIdx.x * 256 + threadIdx.x;
  if (e < Esz) {
    float s = bias[e];
    const float* row = w + (size_t)e * Esz;
    for (int j = 0; j < Esz; j++) s += row[j];
    qv[e] = s;
  }
}

// ---------------------------------------------------------------------------
// Final pooling attention: query length 1, per (b,h). Tiny -> scalar kernel.
// ---------------------------------------------------------------------------
__global__ __launch_bounds__(64) void fin_attn(
    const __bf16* __restrict__ kv, const float* __restrict__ qv,
    const int* __restrict__ mask, float* __restrict__ fa) {
  __shared__ float s[Ssz];
  __shared__ float qh[DHsz];
  __shared__ float red[64];
  int b = blockIdx.x / Hsz;
  int h = blockIdx.x % Hsz;
  int t = threadIdx.x;
  qh[t] = qv[h * DHsz + t];
  __syncthreads();
  for (int k = t; k < Ssz; k += 64) {
    const __bf16* kp = kv + ((size_t)(b * Ssz + k)) * QKVN + Esz + h * DHsz;
    float acc = 0.f;
    for (int d = 0; d < DHsz; d++) acc += qh[d] * (float)kp[d];
    s[k] = acc * 0.125f + (float)mask[b * Ssz + k];
  }
  __syncthreads();
  float mx = -1e30f;
  for (int k = t; k < Ssz; k += 64) mx = fmaxf(mx, s[k]);
  red[t] = mx;
  __syncthreads();
  if (t == 0) { float m = red[0]; for (int j = 1; j < 64; j++) m = fmaxf(m, red[j]); red[0] = m; }
  __syncthreads();
  mx = red[0];
  __syncthreads();
  float sm = 0.f;
  for (int k = t; k < Ssz; k += 64) sm += __expf(s[k] - mx);
  red[t] = sm;
  __syncthreads();
  if (t == 0) { float tt = 0.f; for (int j = 0; j < 64; j++) tt += red[j]; red[0] = tt; }
  __syncthreads();
  float inv = 1.0f / red[0];
  const __bf16* vp = kv + ((size_t)(b * Ssz)) * QKVN + 2 * Esz + h * DHsz + t;
  float o = 0.f;
  for (int k = 0; k < Ssz; k++) o += __expf(s[k] - mx) * (float)vp[(size_t)k * QKVN];
  fa[b * Esz + h * DHsz + t] = o * inv;
}

// pooled[b] = relu(fa[b] @ Wo^T + bo) @ Wl^T + bl  (8 rows only -> scalar)
__global__ __launch_bounds__(256) void pooled_kernel(
    const float* __restrict__ fa, const float* __restrict__ wo,
    const float* __restrict__ bo, const float* __restrict__ wl,
    const float* __restrict__ bl, float* __restrict__ out) {
  int b = blockIdx.x;
  int t = threadIdx.x;
  __shared__ float x[Esz];
  __shared__ float tb[Esz];
  for (int e = t; e < Esz; e += 256) x[e] = fa[b * Esz + e];
  __syncthreads();
  for (int e = t; e < Esz; e += 256) {
    float s = bo[e];
    const float* row = wo + (size_t)e * Esz;
    for (int j = 0; j < Esz; j++) s += row[j] * x[j];
    tb[e] = fmaxf(s, 0.0f);
  }
  __syncthreads();
  for (int e = t; e < Esz; e += 256) {
    float s = bl[e];
    const float* row = wl + (size_t)e * Esz;
    for (int j = 0; j < Esz; j++) s += row[j] * tb[j];
    out[b * Esz + e] = s;
  }
}

// ---------------------------------------------------------------------------
// Host orchestration
// ---------------------------------------------------------------------------
extern "C" void kernel_launch(void* const* d_in, const int* in_sizes, int n_in,
                              void* d_out, int out_size, void* d_ws, size_t ws_size,
                              hipStream_t stream) {
  const float* frozen    = (const float*)d_in[0];
  const int*   mask      = (const int*)d_in[1];
  const float* in_w      = (const float*)d_in[2];
  const float* in_b      = (const float*)d_in[3];
  const float* out_w     = (const float*)d_in[4];
  const float* out_b     = (const float*)d_in[5];
  const float* lin_w     = (const float*)d_in[6];
  const float* lin_b     = (const float*)d_in[7];
  const float* fin_in_w  = (const float*)d_in[8];
  const float* fin_in_b  = (const float*)d_in[9];
  const float* fin_out_w = (const float*)d_in[10];
  const float* fin_out_b = (const float*)d_in[11];
  const float* fin_lin_w = (const float*)d_in[12];
  const float* fin_lin_b = (const float*)d_in[13];
  float* out = (float*)d_out;

  char* ws = (char*)d_ws;
  size_t off = 0;
  auto take = [&](size_t bytes) -> char* {
    char* p = ws + off;
    off = (off + bytes + 255) & ~(size_t)255;
    return p;
  };
  const size_t ACT = (size_t)Bsz * Ssz * Esz;   // 3,145,728
  const size_t WIN = (size_t)QKVN * Esz;        // 1,769,472
  const size_t WEE = (size_t)Esz * Esz;         //   589,824

  __bf16* xb    = (__bf16*)take(ACT * 2);
  __bf16* wib   = (__bf16*)take(WIN * 2);
  __bf16* wob   = (__bf16*)take(WEE * 2);
  __bf16* wlb   = (__bf16*)take(WEE * 2);
  __bf16* qkvb  = (__bf16*)take((size_t)Bsz * Ssz * QKVN * 2);  // reused for fin kv
  __bf16* ctxb  = (__bf16*)take(ACT * 2);
  __bf16* rbuf  = (__bf16*)take(ACT * 2);
  float*  accf  = (float*)take(ACT * 4);
  __bf16* seqb  = (__bf16*)take(ACT * 2);
  __bf16* rseqb = (__bf16*)take(ACT * 2);
  __bf16* flwb  = (__bf16*)take(WEE * 2);
  __bf16* fiwb  = (__bf16*)take(WIN * 2);
  float*  qv    = (float*)take(Esz * 4);
  float*  fa    = (float*)take((size_t)Bsz * Esz * 4);

  auto cast = [&](const float* s, __bf16* d, size_t n) {
    cast_bf16<<<dim3((unsigned)((n + 255) / 256)), dim3(256), 0, stream>>>(s, d,
                                                                           (int)n);
  };

  cast(fin_lin_w, flwb, WEE);
  cast(fin_in_w, fiwb, WIN);

  for (int l = 0; l < LAYERS; l++) {
    cast(frozen + (size_t)l * ACT, xb, ACT);
    cast(in_w + (size_t)l * WIN, wib, WIN);
    cast(out_w + (size_t)l * WEE, wob, WEE);
    cast(lin_w + (size_t)l * WEE, wlb, WEE);
    // qkv = x @ Win^T + b  (bf16 out)
    gemm_wmma<4><<<dim3(64, 36), 128, 0, stream>>>(xb, wib, in_b + (size_t)l * QKVN,
                                                   nullptr, qkvb, 4096, QKVN, Esz);
    // fused MHA core -> bf16 context
    attn_wmma<<<dim3(Bsz * Hsz * 16), 32, 0, stream>>>(qkvb, mask, ctxb);
    // a = relu(ctx @ Wout^T + bo)  (bf16 out, relu)
    gemm_wmma<4 | 1><<<dim3(64, 12), 128, 0, stream>>>(
        ctxb, wob, out_b + (size_t)l * Esz, nullptr, rbuf, 4096, Esz, Esz);
    // acc (+)= a @ Wlin^T + bl  (fp32 out; overwrite on layer 0)
    if (l == 0)
      gemm_wmma<0><<<dim3(64, 12), 128, 0, stream>>>(
          rbuf, wlb, lin_b + (size_t)l * Esz, accf, nullptr, 4096, Esz, Esz);
    else
      gemm_wmma<2><<<dim3(64, 12), 128, 0, stream>>>(
          rbuf, wlb, lin_b + (size_t)l * Esz, accf, nullptr, 4096, Esz, Esz);
  }

  seq_kernel<<<dim3((unsigned)((ACT + 255) / 256)), 256, 0, stream>>>(accf, seqb,
                                                                      rseqb, (int)ACT);
  // unpooled = relu(seq) @ fin_lin^T + b  -> d_out[6144:]
  gemm_wmma<0><<<dim3(64, 12), 128, 0, stream>>>(rseqb, flwb, fin_lin_b,
                                                 out + (size_t)Bsz * Esz, nullptr,
                                                 4096, Esz, Esz);
  // fin k,v = seq @ fin_in^T + b  (bf16, reuse qkv buffer; q-part unused)
  gemm_wmma<4><<<dim3(64, 36), 128, 0, stream>>>(seqb, fiwb, fin_in_b, nullptr, qkvb,
                                                 4096, QKVN, Esz);
  qvec_kernel<<<dim3(3), 256, 0, stream>>>(fin_in_w, fin_in_b, qv);
  fin_attn<<<dim3(Bsz * Hsz), 64, 0, stream>>>(qkvb, qv, mask, fa);
  pooled_kernel<<<dim3(Bsz), 256, 0, stream>>>(fa, fin_out_w, fin_out_b, fin_lin_w,
                                               fin_lin_b, out);
}
